// GuMoeBlock_37778532336142
// MI455X (gfx1250) — matrix operand order repaired and checked by reading, not compiled, not run on hardware
//
#include <hip/hip_runtime.h>
#include <hip/hip_bf16.h>

#define E_ 8
#define H_ 1024
#define I_ 2816
#define K_ 2
#define T_ 2048

#define MT 32                 // tokens per block (2 WMMA M-tiles)
#define NC 128                // I-dimension chunk held in LDS
#define XS_STRIDE (H_ + 8)    // bf16 elems; +8 keeps 16-row frag loads conflict-free
#define ACT_STRIDE (NC + 8)

typedef __attribute__((ext_vector_type(16))) __bf16 v16bf;
typedef __attribute__((ext_vector_type(8)))  float  v8f;

union Frag {
  v16bf v;
  uint4 q[2];
};

__device__ __forceinline__ unsigned short f2bf(float f) {
  unsigned u = __float_as_uint(f);
  u += 0x7FFFu + ((u >> 16) & 1u);   // round-to-nearest-even
  return (unsigned short)(u >> 16);
}

__device__ __forceinline__ v8f wmma_bf16(const Frag& a, const Frag& b, v8f c) {
  return __builtin_amdgcn_wmma_f32_16x16x32_bf16(false, a.v, false, b.v,
                                                 (short)0, c, false, false);
}

// ---------------------------------------------------------------------------
// Kernel 1: deterministic routing lists. Thread e walks tokens in order,
// folding duplicate top-k hits on the same expert into one weight.
// ---------------------------------------------------------------------------
__global__ void route_kernel(const int* __restrict__ idx,
                             const float* __restrict__ val,
                             int* __restrict__ toks, float* __restrict__ wgts,
                             int* __restrict__ counts) {
  int e = threadIdx.x;
  if (e >= E_) return;
  int cnt = 0;
  for (int t = 0; t < T_; ++t) {
    float w = 0.f;
    int hit = 0;
    for (int k = 0; k < K_; ++k) {
      if (idx[t * K_ + k] == e) { w += val[t * K_ + k]; hit = 1; }
    }
    if (hit) {
      toks[e * T_ + cnt] = t;
      wgts[e * T_ + cnt] = w;
      ++cnt;
    }
  }
  counts[e] = cnt;
}

// ---------------------------------------------------------------------------
// Kernel 2: fp32 -> bf16 transpose-convert, src [R][C] -> dst [C][R].
// Col-major weights make WMMA B fragments two contiguous 16B global loads per lane.
// ---------------------------------------------------------------------------
__global__ void tconv_kernel(const float* __restrict__ src,
                             unsigned short* __restrict__ dst,
                             int R, int C) {
  __shared__ float tile[32][33];
  const float* s = src + (size_t)blockIdx.z * R * C;
  unsigned short* d = dst + (size_t)blockIdx.z * R * C;
  int c0 = blockIdx.x * 32, r0 = blockIdx.y * 32;
  int tx = threadIdx.x & 31, ty = threadIdx.x >> 5;
  for (int r = ty; r < 32; r += 8)
    tile[r][tx] = s[(size_t)(r0 + r) * C + c0 + tx];
  __syncthreads();
  for (int c = ty; c < 32; c += 8)
    d[(size_t)(c0 + c) * R + r0 + tx] = f2bf(tile[tx][c]);
}

// ---------------------------------------------------------------------------
// Kernel 3: fused SwiGLU FFN for one (expert, 32-token tile).
//   stage1: G = X@Wg, U = X@Wu (K=1024) per 128-wide I chunk
//   act    = silu(G)*U * routing_weight  -> LDS (bf16)
//   stage2: OUT += act @ Wd (accumulated in VGPRs across chunks)
//   finally atomicAdd into d_out (<=2 experts/token, commutative => determ.)
// ---------------------------------------------------------------------------
__global__ __launch_bounds__(256, 1) void moe_kernel(
    const float* __restrict__ x,
    const unsigned short* __restrict__ wg_t,   // [E][I][H] bf16
    const unsigned short* __restrict__ wu_t,   // [E][I][H] bf16
    const unsigned short* __restrict__ wd_t,   // [E][H][I] bf16
    const int* __restrict__ toks,
    const float* __restrict__ wgts,
    const int* __restrict__ counts,
    float* __restrict__ out) {
  __shared__ unsigned short Xs[MT * XS_STRIDE];   // 66,048 B
  __shared__ unsigned short Act[MT * ACT_STRIDE]; //  8,704 B
  __shared__ float Wsm[MT];
  __shared__ int   Tsm[MT];

  const int e = blockIdx.y;
  const int cnt = counts[e];
  const int base = blockIdx.x * MT;
  if (base >= cnt) return;

  const int tid  = threadIdx.x;
  const int lane = tid & 31;
  const int wid  = tid >> 5;       // 8 waves
  const int l16  = lane & 15;
  const int hh   = lane >> 4;      // lane half

  if (tid < MT) {
    int p = base + tid;
    if (p < cnt) { Tsm[tid] = toks[e * T_ + p]; Wsm[tid] = wgts[e * T_ + p]; }
    else         { Tsm[tid] = toks[e * T_];     Wsm[tid] = 0.f; }   // padding row
  }
  __syncthreads();

  // ---- gather + convert token rows into LDS (bf16) ----
  {
    int row = tid >> 3;                       // 32 rows, 8 threads each
    int cb  = (tid & 7) * (H_ / 8);           // 128 cols per thread
    const float* xr = x + (size_t)Tsm[row] * H_ + cb;
    unsigned short* xd = Xs + row * XS_STRIDE + cb;
    for (int i = 0; i < H_ / 8; i += 4) {
      float4 f = *(const float4*)(xr + i);
      uint2 pk;
      pk.x = (unsigned)f2bf(f.x) | ((unsigned)f2bf(f.y) << 16);
      pk.y = (unsigned)f2bf(f.z) | ((unsigned)f2bf(f.w) << 16);
      *(uint2*)(xd + i) = pk;
    }
  }

  // per-lane routing weights / token ids for the C-fragment rows it owns
  float wr[2][8];
  int   tr[2][8];
#pragma unroll
  for (int mt = 0; mt < 2; ++mt)
#pragma unroll
    for (int j = 0; j < 8; ++j) {
      int r = mt * 16 + hh * 8 + j;
      wr[mt][j] = Wsm[r];
      tr[mt][j] = Tsm[r];
    }
  __syncthreads();

  const unsigned short* wgB = wg_t + (size_t)e * I_ * H_;
  const unsigned short* wuB = wu_t + (size_t)e * I_ * H_;
  const unsigned short* wdB = wd_t + (size_t)e * H_ * I_;

  // stage-2 accumulators: this wave owns out cols [wid*128, wid*128+128)
  const v8f vz = {0.f, 0.f, 0.f, 0.f, 0.f, 0.f, 0.f, 0.f};
  v8f acc[2][8];
#pragma unroll
  for (int mt = 0; mt < 2; ++mt)
#pragma unroll
    for (int nt = 0; nt < 8; ++nt) acc[mt][nt] = vz;

  for (int ic = 0; ic < I_ / NC; ++ic) {
    // ---------------- stage 1: G,U for this wave's 16 I-columns ----------
    v8f g0 = vz, g1 = vz, u0 = vz, u1 = vz;
    const int ncol = ic * NC + wid * 16 + l16;               // global I col
    const unsigned short* wgC = wgB + (size_t)ncol * H_ + hh * 8;
    const unsigned short* wuC = wuB + (size_t)ncol * H_ + hh * 8;
    for (int k0 = 0; k0 < H_; k0 += 32) {
      Frag a0, a1, bg, bu;
      const unsigned short* xa = Xs + l16 * XS_STRIDE + k0 + hh * 8;
      a0.q[0] = *(const uint4*)xa;
      a0.q[1] = *(const uint4*)(xa + 16);
      const unsigned short* xb = Xs + (16 + l16) * XS_STRIDE + k0 + hh * 8;
      a1.q[0] = *(const uint4*)xb;
      a1.q[1] = *(const uint4*)(xb + 16);
      bg.q[0] = *(const uint4*)(wgC + k0);
      bg.q[1] = *(const uint4*)(wgC + k0 + 16);
      bu.q[0] = *(const uint4*)(wuC + k0);
      bu.q[1] = *(const uint4*)(wuC + k0 + 16);
      g0 = wmma_bf16(a0, bg, g0);
      g1 = wmma_bf16(a1, bg, g1);
      u0 = wmma_bf16(a0, bu, u0);
      u1 = wmma_bf16(a1, bu, u1);
    }

    __syncthreads();   // previous chunk's stage-2 readers are done with Act

    // ---- act = silu(g) * u * routing_weight -> LDS bf16 ----
    {
      int nloc = wid * 16 + l16;
#pragma unroll
      for (int mt = 0; mt < 2; ++mt) {
        v8f gg = mt ? g1 : g0;
        v8f uu = mt ? u1 : u0;
#pragma unroll
        for (int j = 0; j < 8; ++j) {
          float gv = gg[j];
          float sv = gv / (1.f + __expf(-gv));     // silu
          float av = sv * uu[j] * wr[mt][j];
          Act[(mt * 16 + hh * 8 + j) * ACT_STRIDE + nloc] = f2bf(av);
        }
      }
    }
    __syncthreads();

    // ---------------- stage 2: OUT += act @ Wd over this chunk -----------
    for (int kk = 0; kk < NC; kk += 32) {
      Frag a0, a1;
      const unsigned short* pa = Act + l16 * ACT_STRIDE + kk + hh * 8;
      a0.q[0] = *(const uint4*)pa;
      a0.q[1] = *(const uint4*)(pa + 16);
      const unsigned short* pb = Act + (16 + l16) * ACT_STRIDE + kk + hh * 8;
      a1.q[0] = *(const uint4*)pb;
      a1.q[1] = *(const uint4*)(pb + 16);
#pragma unroll
      for (int nt = 0; nt < 8; ++nt) {
        int ocol = wid * 128 + nt * 16 + l16;
        const unsigned short* pw =
            wdB + (size_t)ocol * I_ + ic * NC + kk + hh * 8;
        Frag b;
        b.q[0] = *(const uint4*)pw;
        b.q[1] = *(const uint4*)(pw + 16);
        acc[0][nt] = wmma_bf16(a0, b, acc[0][nt]);
        acc[1][nt] = wmma_bf16(a1, b, acc[1][nt]);
      }
    }
  }

  // ---- combine: atomicAdd weighted expert output into d_out ----
#pragma unroll
  for (int mt = 0; mt < 2; ++mt)
#pragma unroll
    for (int nt = 0; nt < 8; ++nt)
#pragma unroll
      for (int j = 0; j < 8; ++j) {
        if (wr[mt][j] != 0.f) {
          atomicAdd(out + (size_t)tr[mt][j] * H_ + wid * 128 + nt * 16 + l16,
                    acc[mt][nt][j]);
        }
      }
}

// ---------------------------------------------------------------------------
extern "C" void kernel_launch(void* const* d_in, const int* in_sizes, int n_in,
                              void* d_out, int out_size, void* d_ws,
                              size_t ws_size, hipStream_t stream) {
  (void)in_sizes; (void)n_in; (void)out_size; (void)ws_size;
  const float* x   = (const float*)d_in[0];
  const int*   tki = (const int*)d_in[1];
  const float* tkv = (const float*)d_in[2];
  const float* wg  = (const float*)d_in[3];
  const float* wu  = (const float*)d_in[4];
  const float* wd  = (const float*)d_in[5];
  float* out = (float*)d_out;

  // workspace layout: 3x bf16 weights (138.4 MB) + routing lists
  const size_t welems = (size_t)E_ * I_ * H_;
  unsigned short* wg_t = (unsigned short*)d_ws;
  unsigned short* wu_t = wg_t + welems;
  unsigned short* wd_t = wu_t + welems;
  int*   toks   = (int*)(wd_t + welems);
  float* wgts   = (float*)(toks + E_ * T_);
  int*   counts = (int*)(wgts + E_ * T_);

  hipMemsetAsync(d_out, 0, (size_t)T_ * H_ * sizeof(float), stream);

  route_kernel<<<1, 32, 0, stream>>>(tki, tkv, toks, wgts, counts);

  // w_gate/w_up: [E][H][I] -> [E][I][H];  w_down: [E][I][H] -> [E][H][I]
  tconv_kernel<<<dim3(I_ / 32, H_ / 32, E_), 256, 0, stream>>>(wg, wg_t, H_, I_);
  tconv_kernel<<<dim3(I_ / 32, H_ / 32, E_), 256, 0, stream>>>(wu, wu_t, H_, I_);
  tconv_kernel<<<dim3(H_ / 32, I_ / 32, E_), 256, 0, stream>>>(wd, wd_t, I_, H_);

  moe_kernel<<<dim3(T_ / MT, E_), 256, 0, stream>>>(x, wg_t, wu_t, wd_t, toks,
                                                    wgts, counts, out);
}
